// TimeCondAttention_32366873543065
// MI455X (gfx1250) — matrix operand order repaired and checked by reading, not compiled, not run on hardware
//
#include <hip/hip_runtime.h>
#include <hip/hip_bf16.h>

// ---------------------------------------------------------------------------
// TimeCondAttention for gfx1250 (MI455X, wave32, WMMA 16x16x32 bf16,
// async global->LDS staging of the 128 MiB attention-bias stream).
// B=2, N=1024, DIM=512, HEADS=8, DIM_HEAD=64, TIME_DIM=512, BIAS_DIM=16
// ---------------------------------------------------------------------------

typedef __bf16 bf16_t;
typedef __attribute__((ext_vector_type(16))) __bf16 v16bf;
typedef __attribute__((ext_vector_type(8)))  float  v8f;

union BF16x16 {            // one WMMA A/B operand (16 halves per lane)
  v16bf v;
  uint4 q[2];
};

__device__ __forceinline__ float red_max16(float x) {
#pragma unroll
  for (int m = 1; m <= 8; m <<= 1) x = fmaxf(x, __shfl_xor(x, m, 32));
  return x;
}
__device__ __forceinline__ float red_sum16(float x) {
#pragma unroll
  for (int m = 1; m <= 8; m <<= 1) x += __shfl_xor(x, m, 32);
  return x;
}

// async global -> LDS, 16 bytes per lane, tracked by ASYNCcnt
__device__ __forceinline__ void async_g2l_b128(unsigned lds_off, const void* gaddr) {
  asm volatile("global_load_async_to_lds_b128 %0, %1, off"
               :: "v"(lds_off), "v"((unsigned long long)(uintptr_t)gaddr)
               : "memory");
}
__device__ __forceinline__ void wait_asynccnt_8() {
  asm volatile("s_wait_asynccnt 0x8" ::: "memory");
}
__device__ __forceinline__ void wait_asynccnt_0() {
  asm volatile("s_wait_asynccnt 0x0" ::: "memory");
}

// ---------------------------------------------------------------------------
// K1: t = silu(time) @ Wt^T + bt   ->  tcond[B, 2*DIM]
// ---------------------------------------------------------------------------
__global__ void k_timecond(const float* __restrict__ time,
                           const float* __restrict__ Wt,
                           const float* __restrict__ bt,
                           float* __restrict__ tcond) {
  int o  = blockIdx.x * blockDim.x + threadIdx.x;   // 0..2047
  int b  = o >> 10;
  int oo = o & 1023;
  const float* tp = time + b * 512;
  const float* wp = Wt + oo * 512;
  float acc = 0.f;
  for (int k = 0; k < 512; ++k) {
    float tv = tp[k];
    float s  = tv / (1.f + __expf(-tv));            // SiLU
    acc += s * wp[k];
  }
  tcond[o] = acc + bt[oo];
}

// ---------------------------------------------------------------------------
// K2: LayerNorm * gamma, * (scale+1) + shift, * seq_mask  -> xn (bf16)
// ---------------------------------------------------------------------------
__global__ void k_lnmod(const float* __restrict__ x,
                        const float* __restrict__ gamma,
                        const float* __restrict__ tcond,
                        const float* __restrict__ seq_mask,
                        bf16_t* __restrict__ xn) {
  __shared__ float red[256];
  int tok = blockIdx.x;          // 0..2047
  int b   = tok >> 10;
  int t   = threadIdx.x;
  const float* xp = x + (size_t)tok * 512;
  float v0 = xp[t], v1 = xp[t + 256];

  red[t] = v0 + v1;
  __syncthreads();
  for (int off = 128; off > 0; off >>= 1) {
    if (t < off) red[t] += red[t + off];
    __syncthreads();
  }
  float mu = red[0] * (1.f / 512.f);
  __syncthreads();

  float d0 = v0 - mu, d1 = v1 - mu;
  red[t] = d0 * d0 + d1 * d1;
  __syncthreads();
  for (int off = 128; off > 0; off >>= 1) {
    if (t < off) red[t] += red[t + off];
    __syncthreads();
  }
  float var  = red[0] * (1.f / 512.f);
  float rstd = rsqrtf(var + 1e-5f);
  float m    = seq_mask[tok];

  const float* sc = tcond + (size_t)b * 1024;
  const float* sh = sc + 512;
#pragma unroll
  for (int e2 = 0; e2 < 2; ++e2) {
    int   e = t + e2 * 256;
    float d = (e2 == 0) ? d0 : d1;
    float r = (d * rstd * gamma[e] * (sc[e] + 1.f) + sh[e]) * m;
    xn[(size_t)tok * 512 + e] = (bf16_t)r;
  }
}

// ---------------------------------------------------------------------------
// K3: f32 -> bf16 weight conversion
// ---------------------------------------------------------------------------
__global__ void k_cvt(const float* __restrict__ src, bf16_t* __restrict__ dst, int n) {
  int i = blockIdx.x * blockDim.x + threadIdx.x;
  if (i < n) dst[i] = (bf16_t)src[i];
}

// ---------------------------------------------------------------------------
// K4: fused QKV projection, WMMA bf16, software-pipelined k-loop.
// C[2048, 1536]: cols 0..511 -> q (scaled, [b,h,n,d]), 512..1023 -> k [b,h,n,d],
// 1024..1535 -> v stored TRANSPOSED [b,h,d,n] for the PV B-tiles.
// ---------------------------------------------------------------------------
__global__ void __launch_bounds__(256) k_qkv(const bf16_t* __restrict__ xn,
                                             const bf16_t* __restrict__ wq,
                                             const bf16_t* __restrict__ wkv,
                                             bf16_t* __restrict__ qb,
                                             bf16_t* __restrict__ kb,
                                             bf16_t* __restrict__ vtb) {
  int wgl    = (blockIdx.x * blockDim.x + threadIdx.x) >> 5;  // global wave id
  int lane   = threadIdx.x & 31;
  int mtile  = wgl / 24;   // 0..127
  int nstrip = wgl % 24;   // 0..23
  int m0     = mtile * 16;
  int hi  = lane >> 4;
  int col = lane & 15;

  v8f acc[4];
#pragma unroll
  for (int jb = 0; jb < 4; ++jb)
#pragma unroll
    for (int e = 0; e < 8; ++e) acc[jb][e] = 0.f;

  BF16x16 a[2], bm[2][4];
  auto load_ab = [&](int k0, int p) {
    const bf16_t* ap = xn + (size_t)(m0 + col) * 512 + k0 + hi * 16;
    a[p].q[0] = *(const uint4*)ap;
    a[p].q[1] = *(const uint4*)(ap + 8);
#pragma unroll
    for (int jb = 0; jb < 4; ++jb) {
      int nb = nstrip * 64 + jb * 16;   // strip-uniform
      const bf16_t* wrow = (nb < 512) ? (wq + (size_t)(nb + col) * 512)
                                      : (wkv + (size_t)(nb - 512 + col) * 512);
      const bf16_t* bp = wrow + k0 + hi * 16;
      bm[p][jb].q[0] = *(const uint4*)bp;
      bm[p][jb].q[1] = *(const uint4*)(bp + 8);
    }
  };

  load_ab(0, 0);
  int p = 0;
  for (int k0 = 0; k0 < 512; k0 += 32) {
    if (k0 + 32 < 512) load_ab(k0 + 32, p ^ 1);   // prefetch next k-step
#pragma unroll
    for (int jb = 0; jb < 4; ++jb)
      acc[jb] = __builtin_amdgcn_wmma_f32_16x16x32_bf16(
          false, a[p].v, false, bm[p][jb].v, (short)0, acc[jb], false, false);
    p ^= 1;
  }

#pragma unroll
  for (int jb = 0; jb < 4; ++jb) {
    int no = nstrip * 64 + jb * 16 + col;
#pragma unroll
    for (int v = 0; v < 8; ++v) {
      int m = m0 + v + hi * 8;          // token
      int b = m >> 10, i = m & 1023;
      float r = acc[jb][v];
      if (no < 512) {                   // q, pre-scaled by d^-0.5
        int h = no >> 6, d = no & 63;
        qb[((size_t)(b * 8 + h) * 1024 + i) * 64 + d] = (bf16_t)(r * 0.125f);
      } else if (no < 1024) {           // k
        int n2 = no - 512, h = n2 >> 6, d = n2 & 63;
        kb[((size_t)(b * 8 + h) * 1024 + i) * 64 + d] = (bf16_t)r;
      } else {                          // v transposed: [b,h,d,n]
        int n3 = no - 1024, h = n3 >> 6, d = n3 & 63;
        vtb[((size_t)(b * 8 + h) * 64 + d) * 1024 + i] = (bf16_t)r;
      }
    }
  }
}

// ---------------------------------------------------------------------------
// K5: flash attention, 8 heads per workgroup (wave w = head w) sharing one
// 16-row query tile so the 128 MiB attn_bias is read from HBM exactly once.
// Per j-step: (1) all K/V tiles issued up-front so the V latency hides under
// the softmax VALU work, (2) the NEXT step's raw bias tile [16a][16i][32j]
// (32 KB) is staged with GLOBAL_LOAD_ASYNC_TO_LDS_B128 into a ping-pong LDS
// buffer, (3) 4 WMMAs QK^T + online softmax + 4 WMMAs PV per wave.
// ---------------------------------------------------------------------------
__global__ void __launch_bounds__(256) k_attn(const bf16_t* __restrict__ qb,
                                              const bf16_t* __restrict__ kb,
                                              const bf16_t* __restrict__ vtb,
                                              const float* __restrict__ attn_bias,
                                              const float* __restrict__ Wb,
                                              const float* __restrict__ bbias,
                                              const float* __restrict__ seq_mask,
                                              bf16_t* __restrict__ ob) {
  __shared__ float s_bias[2][16 * 16 * 32];            // 2 x 32 KB ping-pong
  __shared__ __align__(16) bf16_t s_p[8][16 * 32];     // 8 KB per-wave P tiles

  int b    = blockIdx.y;
  int i0   = blockIdx.x * 16;
  int wv   = threadIdx.x >> 5;     // wave = head
  int lane = threadIdx.x & 31;
  int hi   = lane >> 4;
  int col  = lane & 15;
  int h    = wv;
  int bh   = b * 8 + h;

  // q tile (16 rows x 64) held in registers for the whole j loop
  BF16x16 aq[2];
#pragma unroll
  for (int ks = 0; ks < 2; ++ks) {
    const bf16_t* ap = qb + ((size_t)bh * 1024 + i0 + col) * 64 + ks * 32 + hi * 16;
    aq[ks].q[0] = *(const uint4*)ap;
    aq[ks].q[1] = *(const uint4*)(ap + 8);
  }
  float wbv[16];
#pragma unroll
  for (int a = 0; a < 16; ++a) wbv[a] = Wb[h * 16 + a];
  float bias0 = bbias[h];

  float mrow[8], lrow[8];
  v8f acc[4];
#pragma unroll
  for (int v = 0; v < 8; ++v) { mrow[v] = -1e30f; lrow[v] = 0.f; }
#pragma unroll
  for (int dk = 0; dk < 4; ++dk)
#pragma unroll
    for (int e = 0; e < 8; ++e) acc[dk][e] = 0.f;

  // issue one 32 KB bias tile (8 x b128 per thread) asynchronously into LDS
  auto issue_bias_tile = [&](int j0, int buf) {
    int t = threadIdx.x;
#pragma unroll
    for (int kk = 0; kk < 8; ++kk) {
      int f   = (t + kk * 256) * 4;          // flat f32 index, 16B aligned
      int a   = f >> 9;
      int rem = f & 511;
      int ii  = rem >> 5;
      int jj  = rem & 31;
      const float* gp = attn_bias +
          ((size_t)(b * 16 + a) * 1024 + (i0 + ii)) * 1024 + j0 + jj;
      unsigned lds_off = (unsigned)(uintptr_t)(const void*)&s_bias[buf][f];
      async_g2l_b128(lds_off, gp);
    }
  };

  issue_bias_tile(0, 0);
  int pb = 0;
  for (int j0 = 0; j0 < 1024; j0 += 32) {
    // ---- issue ALL K/V tile loads for this j-step up front ----
    BF16x16 kt[4];                        // [jb*2+ks]
#pragma unroll
    for (int jb = 0; jb < 2; ++jb)
#pragma unroll
      for (int ks = 0; ks < 2; ++ks) {
        const bf16_t* bp =
            kb + ((size_t)bh * 1024 + j0 + jb * 16 + col) * 64 + ks * 32 + hi * 16;
        kt[jb * 2 + ks].q[0] = *(const uint4*)bp;
        kt[jb * 2 + ks].q[1] = *(const uint4*)(bp + 8);
      }
    BF16x16 vt[4];                        // consumed only after softmax
#pragma unroll
    for (int dk = 0; dk < 4; ++dk) {
      const bf16_t* vp =
          vtb + ((size_t)bh * 64 + dk * 16 + col) * 1024 + j0 + hi * 16;
      vt[dk].q[0] = *(const uint4*)vp;
      vt[dk].q[1] = *(const uint4*)(vp + 8);
    }

    if (j0 + 32 < 1024) {
      issue_bias_tile(j0 + 32, pb ^ 1);   // prefetch next tile
      wait_asynccnt_8();                  // previous tile complete (in-order)
    } else {
      wait_asynccnt_0();
    }
    __syncthreads();
    const float* bias_t = s_bias[pb];

    // ---- sim = q k^T (+ projected bias + col mask) ----
    v8f sim[2];
#pragma unroll
    for (int jb = 0; jb < 2; ++jb) {
#pragma unroll
      for (int e = 0; e < 8; ++e) sim[jb][e] = 0.f;
#pragma unroll
      for (int ks = 0; ks < 2; ++ks)
        sim[jb] = __builtin_amdgcn_wmma_f32_16x16x32_bf16(
            false, aq[ks].v, false, kt[jb * 2 + ks].v, (short)0, sim[jb],
            false, false);
      int   j  = j0 + jb * 16 + col;
      float mj = seq_mask[b * 1024 + j];
      float madd = (mj - 1.f) * 1e6f;
#pragma unroll
      for (int v = 0; v < 8; ++v) {
        int   il = v + hi * 8;
        float bs = bias0;
#pragma unroll
        for (int a = 0; a < 16; ++a)
          bs += bias_t[(a * 16 + il) * 32 + jb * 16 + col] * wbv[a];
        sim[jb][v] += bs + madd;
      }
    }

    // ---- online softmax (rows live across 16-lane halves) ----
    // V-tile loads are still in flight here, hidden under this VALU work.
#pragma unroll
    for (int v = 0; v < 8; ++v) {
      float cmax = red_max16(fmaxf(sim[0][v], sim[1][v]));
      float mnew = fmaxf(mrow[v], cmax);
      float scl  = __expf(mrow[v] - mnew);
      float p0   = __expf(sim[0][v] - mnew);
      float p1   = __expf(sim[1][v] - mnew);
      float rs   = red_sum16(p0 + p1);
      lrow[v] = lrow[v] * scl + rs;
      mrow[v] = mnew;
#pragma unroll
      for (int dk = 0; dk < 4; ++dk) acc[dk][v] *= scl;
      int il = v + hi * 8;
      s_p[wv][il * 32 + col]      = (bf16_t)p0;
      s_p[wv][il * 32 + 16 + col] = (bf16_t)p1;
    }

    // ---- out += P V : P re-read from LDS in A layout, V^T already in regs ----
    BF16x16 ap;
    const bf16_t* pp = &s_p[wv][col * 32 + hi * 16];
    ap.q[0] = *(const uint4*)pp;
    ap.q[1] = *(const uint4*)(pp + 8);
#pragma unroll
    for (int dk = 0; dk < 4; ++dk)
      acc[dk] = __builtin_amdgcn_wmma_f32_16x16x32_bf16(
          false, ap.v, false, vt[dk].v, (short)0, acc[dk], false, false);

    __syncthreads();   // all waves done reading bias_t before it is re-filled
    pb ^= 1;
  }

  // ---- epilogue: normalize, write [b, n, h*d] bf16 ----
#pragma unroll
  for (int v = 0; v < 8; ++v) {
    float inv = 1.f / fmaxf(lrow[v], 1e-20f);
    int   i   = i0 + v + hi * 8;
#pragma unroll
    for (int dk = 0; dk < 4; ++dk) {
      ob[(size_t)(b * 1024 + i) * 512 + h * 64 + dk * 16 + col] =
          (bf16_t)(acc[dk][v] * inv);
    }
  }
}

// ---------------------------------------------------------------------------
// K6: out = (attn_out @ Wo^T) * seq_mask, WMMA bf16, pipelined, f32 output
// ---------------------------------------------------------------------------
__global__ void __launch_bounds__(256) k_oproj(const bf16_t* __restrict__ ao,
                                               const bf16_t* __restrict__ wo,
                                               const float* __restrict__ seq_mask,
                                               float* __restrict__ out) {
  int wgl    = (blockIdx.x * blockDim.x + threadIdx.x) >> 5;
  int lane   = threadIdx.x & 31;
  int mtile  = wgl >> 3;    // 0..127
  int nstrip = wgl & 7;     // 0..7
  int m0  = mtile * 16;
  int hi  = lane >> 4;
  int col = lane & 15;

  v8f acc[4];
#pragma unroll
  for (int jb = 0; jb < 4; ++jb)
#pragma unroll
    for (int e = 0; e < 8; ++e) acc[jb][e] = 0.f;

  BF16x16 a[2], bm[2][4];
  auto load_ab = [&](int k0, int p) {
    const bf16_t* ap = ao + (size_t)(m0 + col) * 512 + k0 + hi * 16;
    a[p].q[0] = *(const uint4*)ap;
    a[p].q[1] = *(const uint4*)(ap + 8);
#pragma unroll
    for (int jb = 0; jb < 4; ++jb) {
      int no = nstrip * 64 + jb * 16 + col;
      const bf16_t* bp = wo + (size_t)no * 512 + k0 + hi * 16;
      bm[p][jb].q[0] = *(const uint4*)bp;
      bm[p][jb].q[1] = *(const uint4*)(bp + 8);
    }
  };

  load_ab(0, 0);
  int p = 0;
  for (int k0 = 0; k0 < 512; k0 += 32) {
    if (k0 + 32 < 512) load_ab(k0 + 32, p ^ 1);
#pragma unroll
    for (int jb = 0; jb < 4; ++jb)
      acc[jb] = __builtin_amdgcn_wmma_f32_16x16x32_bf16(
          false, a[p].v, false, bm[p][jb].v, (short)0, acc[jb], false, false);
    p ^= 1;
  }

#pragma unroll
  for (int jb = 0; jb < 4; ++jb) {
    int no = nstrip * 64 + jb * 16 + col;
#pragma unroll
    for (int v = 0; v < 8; ++v) {
      int m = m0 + v + hi * 8;
      out[(size_t)m * 512 + no] = acc[jb][v] * seq_mask[m];
    }
  }
}

// ---------------------------------------------------------------------------
extern "C" void kernel_launch(void* const* d_in, const int* in_sizes, int n_in,
                              void* d_out, int out_size, void* d_ws, size_t ws_size,
                              hipStream_t stream) {
  const float* x         = (const float*)d_in[0];
  // d_in[1] residx: unused by the reference math
  const float* time      = (const float*)d_in[2];
  const float* attn_bias = (const float*)d_in[3];
  const float* seq_mask  = (const float*)d_in[4];
  const float* gamma     = (const float*)d_in[5];
  const float* Wt        = (const float*)d_in[6];
  const float* bt        = (const float*)d_in[7];
  const float* Wb        = (const float*)d_in[8];
  const float* bb        = (const float*)d_in[9];
  const float* Wq        = (const float*)d_in[10];
  const float* Wkv       = (const float*)d_in[11];
  const float* Wo        = (const float*)d_in[12];

  char* ws = (char*)d_ws;
  size_t off = 0;
  auto alloc = [&](size_t bytes) { void* p = ws + off; off = (off + bytes + 255) & ~(size_t)255; return p; };

  float*  tcond  = (float*)alloc(2 * 1024 * sizeof(float));
  bf16_t* xn     = (bf16_t*)alloc((size_t)2048 * 512 * 2);
  bf16_t* wq_bf  = (bf16_t*)alloc((size_t)512 * 512 * 2);
  bf16_t* wkv_bf = (bf16_t*)alloc((size_t)1024 * 512 * 2);
  bf16_t* wo_bf  = (bf16_t*)alloc((size_t)512 * 512 * 2);
  bf16_t* qb     = (bf16_t*)alloc((size_t)2 * 8 * 1024 * 64 * 2);
  bf16_t* kbuf   = (bf16_t*)alloc((size_t)2 * 8 * 1024 * 64 * 2);
  bf16_t* vtb    = (bf16_t*)alloc((size_t)2 * 8 * 64 * 1024 * 2);
  bf16_t* aout   = (bf16_t*)alloc((size_t)2048 * 512 * 2);
  (void)ws_size; (void)in_sizes; (void)n_in; (void)out_size;

  k_timecond<<<8, 256, 0, stream>>>(time, Wt, bt, tcond);
  k_lnmod<<<2048, 256, 0, stream>>>(x, gamma, tcond, seq_mask, xn);
  k_cvt<<<(512 * 512 + 255) / 256, 256, 0, stream>>>(Wq, wq_bf, 512 * 512);
  k_cvt<<<(1024 * 512 + 255) / 256, 256, 0, stream>>>(Wkv, wkv_bf, 1024 * 512);
  k_cvt<<<(512 * 512 + 255) / 256, 256, 0, stream>>>(Wo, wo_bf, 512 * 512);
  k_qkv<<<384, 256, 0, stream>>>(xn, wq_bf, wkv_bf, qb, kbuf, vtb);
  k_attn<<<dim3(64, 2), 256, 0, stream>>>(qb, kbuf, vtb, attn_bias, Wb, bb,
                                          seq_mask, aout);
  k_oproj<<<128, 256, 0, stream>>>(aout, wo_bf, seq_mask, (float*)d_out);
}
